// decoderAttention_12498354831698
// MI455X (gfx1250) — matrix operand order repaired and checked by reading, not compiled, not run on hardware
//
#include <hip/hip_runtime.h>
#include <cstdint>

// ---------------------------------------------------------------------------
// Problem constants (from reference): B=1, N=4098, C=384, H=6, D=64
// ---------------------------------------------------------------------------
#define CDIM   384
#define NQ     4098
#define NKEY   4099          // N+1 keys (back token appended)
#define NHEAD  6
#define DHEAD  64
#define NINT   4096          // interior tokens N-2
#define MPAD   4112          // queries padded to multiple of 16 (257 tiles)
#define KVPAD  4128          // keys padded to multiple of 32 (129 FA iters)
#define NEGBIG (-1.0e9f)

typedef unsigned short u16;
typedef __bf16 v16bf __attribute__((ext_vector_type(16)));
typedef __bf16 v8bf  __attribute__((ext_vector_type(8)));
typedef float  v8f   __attribute__((ext_vector_type(8)));

union BF16Frag { v16bf v; v8bf h[2]; };
union BF16Half { v8bf v; u16 u[8]; };

__device__ __forceinline__ u16 f2bf(float f) {
  union { float f; unsigned u; } c; c.f = f;
  unsigned x = c.u;
  x += 0x7FFFu + ((x >> 16) & 1u);       // round-to-nearest-even
  return (u16)(x >> 16);
}

#define WMMA_BF16(A, B, Cacc) \
  __builtin_amdgcn_wmma_f32_16x16x32_bf16(false, (A), false, (B), (short)0, (Cacc), false, false)

// ---------------------------------------------------------------------------
// Kernel 1: back token = mean of x[1:N-1] rows with mask < 0.5
// one block per channel; strided reduction over the 4096 interior tokens
// ---------------------------------------------------------------------------
__global__ __launch_bounds__(256) void back_token_kernel(
    const float* __restrict__ x, const float* __restrict__ mask,
    float* __restrict__ back) {
  const int c = blockIdx.x;
  float sum = 0.f, cnt = 0.f;
  for (int n = threadIdx.x; n < NINT; n += 256) {
    float keep = (mask[n] < 0.5f) ? 1.0f : 0.0f;
    sum += keep * x[(size_t)(1 + n) * CDIM + c];
    cnt += keep;
  }
  __shared__ float ssum[256], scnt[256];
  ssum[threadIdx.x] = sum; scnt[threadIdx.x] = cnt;
  __syncthreads();
  for (int s = 128; s > 0; s >>= 1) {
    if (threadIdx.x < s) {
      ssum[threadIdx.x] += ssum[threadIdx.x + s];
      scnt[threadIdx.x] += scnt[threadIdx.x + s];
    }
    __syncthreads();
  }
  if (threadIdx.x == 0) back[c] = ssum[0] / (scnt[0] + 1e-10f);
}

// ---------------------------------------------------------------------------
// Kernel 2: convert 4 weight matrices fp32 -> bf16 (concatenated Wq|Wk|Wv|Wp)
// ---------------------------------------------------------------------------
__global__ __launch_bounds__(256) void cvt_w_kernel(
    const float* __restrict__ Wq, const float* __restrict__ Wk,
    const float* __restrict__ Wv, const float* __restrict__ Wp,
    u16* __restrict__ out) {
  int idx = blockIdx.x * 256 + threadIdx.x;
  const int CC = CDIM * CDIM;
  if (idx >= 4 * CC) return;
  int sel = idx / CC, k = idx % CC;
  const float* src = (sel == 0) ? Wq : (sel == 1) ? Wk : (sel == 2) ? Wv : Wp;
  out[idx] = f2bf(src[k]);
}

// ---------------------------------------------------------------------------
// Kernel 3: pack activations to bf16 with zero padding + build key bias
//   qx [MPAD  x C] = x + pos            (queries)
//   kx [KVPAD x C] = kv_x + pos         (keys)
//   kvx[KVPAD x C] = kv_x               (values source)
//   bias[KVPAD]    = keep ? 0 : -1e9
// ---------------------------------------------------------------------------
__global__ __launch_bounds__(256) void pack_kernel(
    const float* __restrict__ x, const float* __restrict__ pos,
    const float* __restrict__ mask, const float* __restrict__ mask_block,
    const float* __restrict__ back,
    u16* __restrict__ qx, u16* __restrict__ kx, u16* __restrict__ kvx,
    float* __restrict__ bias) {
  int idx = blockIdx.x * 256 + threadIdx.x;
  if (idx >= KVPAD * CDIM) return;
  int j = idx / CDIM, c = idx % CDIM;

  float kv = 0.f, p = 0.f;
  if (j < NQ)       kv = x[(size_t)j * CDIM + c];
  else if (j == NQ) kv = back[c];
  if (j < NKEY)     p  = pos[(size_t)j * CDIM + c];

  kvx[idx] = f2bf((j < NKEY) ? kv : 0.f);
  kx[idx]  = f2bf((j < NKEY) ? (kv + p) : 0.f);
  if (j < MPAD)
    qx[(size_t)j * CDIM + c] = f2bf((j < NQ) ? (x[(size_t)j * CDIM + c] + p) : 0.f);

  if (c == 0) {
    float mbv = (j < NKEY) ? mask_block[j] : 0.f;
    if (j >= 1 && j <= NQ - 2) mbv = mask[j - 1];   // interior override
    bias[j] = (j < NKEY && mbv >= 0.5f) ? 0.f : NEGBIG;
  }
}

// ---------------------------------------------------------------------------
// Kernel 4: generic WMMA projection GEMM  Y = alpha * (A[MxK] @ W[OxK]^T)
//   A bf16 row-major (ld = 384), W bf16 row-major (ld = 384), O = 384
//   one wave -> one 16x16 output tile; 8 waves per block
//   mode 0: bf16 row-major store (ldy = C)
//   mode 1: bf16 transposed store (Vt[o][m], ldy = token stride)
//   mode 2: fp32 row-major store + bias, rows < nvalid only (final output)
// ---------------------------------------------------------------------------
__global__ __launch_bounds__(256) void proj_gemm_kernel(
    const u16* __restrict__ A, const u16* __restrict__ W, void* __restrict__ Y,
    const float* __restrict__ bias, int ntiles, int ldy, float alpha,
    int mode, int nvalid) {
  const int lane = threadIdx.x & 31;
  const int wave = threadIdx.x >> 5;
  const int tile = blockIdx.x * 8 + wave;
  if (tile >= ntiles) return;                 // whole wave exits together
  const int OT = CDIM / 16;                   // 24 output-column tiles
  const int m0 = (tile / OT) * 16;
  const int o0 = (tile % OT) * 16;
  const int half = lane >> 4, r = lane & 15;

  v8f acc = {};
  for (int k0 = 0; k0 < CDIM; k0 += 32) {
    // A fragment: row m0+r; ISA A-layout -> K chunks {half*8..+7, 16+half*8..+7}
    BF16Frag a, b;
    const u16* ap = A + (size_t)(m0 + r) * CDIM + k0;
    a.h[0] = *reinterpret_cast<const v8bf*>(ap + half * 8);
    a.h[1] = *reinterpret_cast<const v8bf*>(ap + 16 + half * 8);
    // B fragment: B[k][o] = W[o][k]; lane column o0+r, K chunks {half*8..+7, 16+half*8..+7}
    const u16* wp = W + (size_t)(o0 + r) * CDIM + k0;
    b.h[0] = *reinterpret_cast<const v8bf*>(wp + half * 8);
    b.h[1] = *reinterpret_cast<const v8bf*>(wp + 16 + half * 8);
    acc = WMMA_BF16(a.v, b.v, acc);
  }

  // C/D layout: VGPR j, lane -> row m0 + j + 8*half, col o0 + r
  if (mode == 0) {
    u16* y = (u16*)Y;
    #pragma unroll
    for (int j = 0; j < 8; ++j)
      y[(size_t)(m0 + 8 * half + j) * ldy + o0 + r] = f2bf(acc[j] * alpha);
  } else if (mode == 1) {
    // transposed: 8 contiguous bf16 along token dim -> one 16B store
    BF16Half hv;
    #pragma unroll
    for (int j = 0; j < 8; ++j) hv.u[j] = f2bf(acc[j] * alpha);
    *reinterpret_cast<v8bf*>((u16*)Y + (size_t)(o0 + r) * ldy + m0 + 8 * half) = hv.v;
  } else {
    float* y = (float*)Y;
    const float bv = bias[o0 + r];
    #pragma unroll
    for (int j = 0; j < 8; ++j) {
      int row = m0 + 8 * half + j;
      if (row < nvalid) y[(size_t)row * ldy + o0 + r] = acc[j] + bv;
    }
  }
}

// ---------------------------------------------------------------------------
// Kernel 5: flash attention, one wave per (16-query tile, head)
//   Qb [MPAD  x C] bf16 (pre-scaled by D^-0.5), Kb [KVPAD x C] bf16,
//   Vt [C x KVPAD] bf16 (transposed), bias [KVPAD], Ob [MPAD x C] bf16
//   32 keys/iter: 4 WMMAs for scores, online softmax, P via LDS, 4 WMMAs P*V
// ---------------------------------------------------------------------------
__global__ __launch_bounds__(32) void attn_kernel(
    const u16* __restrict__ Qb, const u16* __restrict__ Kb,
    const u16* __restrict__ Vt, const float* __restrict__ bias,
    u16* __restrict__ Ob) {
  __shared__ __align__(16) u16 lds_p[16 * 32];   // 16 queries x 32 keys bf16
  const int lane = threadIdx.x;
  const int half = lane >> 4, r = lane & 15;
  const int m0 = blockIdx.x * 16;
  const int h  = blockIdx.y;

  // Q A-fragments for the two 32-wide d-chunks (D=64), loaded once
  BF16Frag aQ[2];
  #pragma unroll
  for (int c = 0; c < 2; ++c) {
    const u16* qp = Qb + (size_t)(m0 + r) * CDIM + h * DHEAD + c * 32;
    aQ[c].h[0] = *reinterpret_cast<const v8bf*>(qp + half * 8);
    aQ[c].h[1] = *reinterpret_cast<const v8bf*>(qp + 16 + half * 8);
  }

  v8f oacc[4] = {{}, {}, {}, {}};
  float mrow[8], lrow[8];
  #pragma unroll
  for (int jj = 0; jj < 8; ++jj) { mrow[jj] = -3.0e38f; lrow[jj] = 0.f; }

  for (int j0 = 0; j0 < KVPAD; j0 += 32) {
    // --- scores: S0 = Q . K[j0..j0+15]^T, S1 = Q . K[j0+16..j0+31]^T ---
    v8f s0 = {}, s1 = {};
    #pragma unroll
    for (int c = 0; c < 2; ++c) {
      BF16Frag bk;
      const u16* kp0 = Kb + (size_t)(j0 + r) * CDIM + h * DHEAD + c * 32;
      bk.h[0] = *reinterpret_cast<const v8bf*>(kp0 + half * 8);
      bk.h[1] = *reinterpret_cast<const v8bf*>(kp0 + 16 + half * 8);
      s0 = WMMA_BF16(aQ[c].v, bk.v, s0);
      const u16* kp1 = Kb + (size_t)(j0 + 16 + r) * CDIM + h * DHEAD + c * 32;
      bk.h[0] = *reinterpret_cast<const v8bf*>(kp1 + half * 8);
      bk.h[1] = *reinterpret_cast<const v8bf*>(kp1 + 16 + half * 8);
      s1 = WMMA_BF16(aQ[c].v, bk.v, s1);
    }
    const float bb0 = bias[j0 + r];
    const float bb1 = bias[j0 + 16 + r];

    // --- online softmax; row q = jj + 8*half lives on 16 lanes (same half) ---
    #pragma unroll
    for (int jj = 0; jj < 8; ++jj) {
      float x0 = s0[jj] + bb0;
      float x1 = s1[jj] + bb1;
      float mx = fmaxf(x0, x1);
      #pragma unroll
      for (int off = 1; off < 16; off <<= 1)
        mx = fmaxf(mx, __shfl_xor(mx, off, 32));
      float mnew = fmaxf(mrow[jj], mx);
      float corr = __expf(mrow[jj] - mnew);
      float e0 = __expf(x0 - mnew);
      float e1 = __expf(x1 - mnew);
      float sum = e0 + e1;
      #pragma unroll
      for (int off = 1; off < 16; off <<= 1)
        sum += __shfl_xor(sum, off, 32);
      lrow[jj] = lrow[jj] * corr + sum;
      mrow[jj] = mnew;
      oacc[0][jj] *= corr; oacc[1][jj] *= corr;
      oacc[2][jj] *= corr; oacc[3][jj] *= corr;
      // stage P (C-layout) into LDS for A-layout reload
      lds_p[(jj + 8 * half) * 32 + r]      = f2bf(e0);
      lds_p[(jj + 8 * half) * 32 + 16 + r] = f2bf(e1);
    }
    __syncthreads();
    BF16Frag aP;
    aP.h[0] = *reinterpret_cast<const v8bf*>(&lds_p[r * 32 + half * 8]);
    aP.h[1] = *reinterpret_cast<const v8bf*>(&lds_p[r * 32 + 16 + half * 8]);
    __syncthreads();

    // --- O += P[16x32] @ V[32 x 16] per d-tile; Vt rows are contiguous in j ---
    #pragma unroll
    for (int dt = 0; dt < 4; ++dt) {
      BF16Frag bv;
      const u16* vp = Vt + (size_t)(h * DHEAD + dt * 16 + r) * KVPAD + j0;
      bv.h[0] = *reinterpret_cast<const v8bf*>(vp + half * 8);
      bv.h[1] = *reinterpret_cast<const v8bf*>(vp + 16 + half * 8);
      oacc[dt] = WMMA_BF16(aP.v, bv.v, oacc[dt]);
    }
  }

  // --- finalize: divide by l, store bf16 row-major for the output GEMM ---
  #pragma unroll
  for (int dt = 0; dt < 4; ++dt) {
    #pragma unroll
    for (int jj = 0; jj < 8; ++jj) {
      float val = oacc[dt][jj] / lrow[jj];
      Ob[(size_t)(m0 + jj + 8 * half) * CDIM + h * DHEAD + dt * 16 + r] = f2bf(val);
    }
  }
}

// ---------------------------------------------------------------------------
// Host-side orchestration
// inputs: x, pos, mask, mask_block, Wq, Wk, Wv, Wp, bp  (all fp32)
// output: (1, N, C) fp32
// ---------------------------------------------------------------------------
extern "C" void kernel_launch(void* const* d_in, const int* in_sizes, int n_in,
                              void* d_out, int out_size, void* d_ws, size_t ws_size,
                              hipStream_t stream) {
  const float* x          = (const float*)d_in[0];
  const float* pos        = (const float*)d_in[1];
  const float* mask       = (const float*)d_in[2];
  const float* mask_block = (const float*)d_in[3];
  const float* Wq         = (const float*)d_in[4];
  const float* Wk         = (const float*)d_in[5];
  const float* Wv         = (const float*)d_in[6];
  const float* Wp         = (const float*)d_in[7];
  const float* bp         = (const float*)d_in[8];

  // workspace carve-out (256B aligned)
  char* ws = (char*)d_ws;
  auto carve = [&](size_t bytes) -> void* {
    void* p = (void*)ws;
    ws += (bytes + 255) & ~(size_t)255;
    return p;
  };
  const size_t CC = (size_t)CDIM * CDIM;
  float* back  = (float*)carve(CDIM * sizeof(float));
  float* bias  = (float*)carve(KVPAD * sizeof(float));
  u16*   Wb    = (u16*)  carve(4 * CC * sizeof(u16));        // Wq|Wk|Wv|Wp bf16
  u16*   qx    = (u16*)  carve((size_t)MPAD  * CDIM * sizeof(u16));
  u16*   kx    = (u16*)  carve((size_t)KVPAD * CDIM * sizeof(u16));
  u16*   kvx   = (u16*)  carve((size_t)KVPAD * CDIM * sizeof(u16));
  u16*   qproj = (u16*)  carve((size_t)MPAD  * CDIM * sizeof(u16));
  u16*   kproj = (u16*)  carve((size_t)KVPAD * CDIM * sizeof(u16));
  u16*   vt    = (u16*)  carve((size_t)CDIM  * KVPAD * sizeof(u16));
  u16*   obuf  = (u16*)  carve((size_t)MPAD  * CDIM * sizeof(u16));

  // 1) back token
  back_token_kernel<<<CDIM, 256, 0, stream>>>(x, mask, back);

  // 2) weights -> bf16
  cvt_w_kernel<<<(int)((4 * CC + 255) / 256), 256, 0, stream>>>(Wq, Wk, Wv, Wp, Wb);

  // 3) pack activations + key bias
  pack_kernel<<<(KVPAD * CDIM + 255) / 256, 256, 0, stream>>>(
      x, pos, mask, mask_block, back, qx, kx, kvx, bias);

  // 4) projections (wave-per-16x16-tile WMMA GEMMs)
  const int TQ = (MPAD  / 16) * (CDIM / 16);   // 6168 tiles
  const int TK = (KVPAD / 16) * (CDIM / 16);   // 6192 tiles
  // Q = (x+pos) Wq^T * D^-0.5  -> bf16 row-major
  proj_gemm_kernel<<<(TQ + 7) / 8, 256, 0, stream>>>(
      qx, Wb + 0 * CC, qproj, nullptr, TQ, CDIM, 0.125f, 0, 0);
  // K = (kv+pos) Wk^T          -> bf16 row-major
  proj_gemm_kernel<<<(TK + 7) / 8, 256, 0, stream>>>(
      kx, Wb + 1 * CC, kproj, nullptr, TK, CDIM, 1.0f, 0, 0);
  // V = kv Wv^T                -> bf16 TRANSPOSED (channel-major, ld = KVPAD)
  proj_gemm_kernel<<<(TK + 7) / 8, 256, 0, stream>>>(
      kvx, Wb + 2 * CC, vt, nullptr, TK, KVPAD, 1.0f, 1, 0);

  // 5) flash attention: one wave per (16-query tile, head)
  attn_kernel<<<dim3(MPAD / 16, NHEAD), 32, 0, stream>>>(qproj, kproj, vt, bias, obuf);

  // 6) output projection + bias -> fp32 d_out (valid rows only)
  proj_gemm_kernel<<<(TQ + 7) / 8, 256, 0, stream>>>(
      obuf, Wb + 3 * CC, d_out, bp, TQ, CDIM, 1.0f, 2, NQ);
}
